// OutputBlock_42004780155368
// MI455X (gfx1250) — matrix operand order repaired and checked by reading, not compiled, not run on hardware
//
#include <hip/hip_runtime.h>
#include <hip/hip_bf16.h>

// ---------------------------------------------------------------------------
// MI455X (gfx1250) implementation.
//
// Stage A (bandwidth-bound, ~430 MB streamed -> ~19us floor @ 23.3 TB/s):
//   gate = rbf @ W_rbf (6x128, per-edge in registers), gated = messages*gate,
//   f64 scatter-add via inline-asm global_atomic_add_f64 (51 MB accumulator
//   lives in the 192 MB L2, absorbing the ~16-way collision rate).
// Stage B (compute, ~23 GFLOP): 50k x (128->256 -> 3x 256x256 swish -> 1)
//   via v_wmma_f32_16x16x32_f16, one wave per 16-row strip (50000 = 3125*16),
//   activations staged in LDS (f16, ds_load/ds_store after full inlining),
//   weights pre-packed in fragment order so each B fragment is a contiguous
//   32B/lane global_load_b128 pair.
// ---------------------------------------------------------------------------

#define N_EDGES   800000
#define N_ATOMS   50000
#define EMBED     128
#define RBF_N     6
#define OUT_EMBED 256
#define STRIPS    (N_ATOMS / 16)          // 3125 exact

typedef __attribute__((ext_vector_type(16))) _Float16 v16h;
typedef __attribute__((ext_vector_type(8)))  float    v8f;

// f64 atomic add without return (STOREcnt-tracked); guarantees the native
// L2 atomic instead of a CAS loop.
__device__ __forceinline__ void atomic_add_f64(double* p, double v) {
    unsigned long long a = (unsigned long long)p;
    asm volatile("global_atomic_add_f64 %0, %1, off" : : "v"(a), "v"(v) : "memory");
}

// ---------------------------------------------------------------------------
// Kernel 1: zero the f64 accumulator (workspace is poisoned / carries state).
// ---------------------------------------------------------------------------
__global__ void zero_acc_kernel(double* __restrict__ acc) {
    size_t i = (size_t)blockIdx.x * blockDim.x + threadIdx.x;   // 6.4M exact
    acc[i] = 0.0;
}

// ---------------------------------------------------------------------------
// Kernel 2: pack W_up (128x256) and W_dense (3x256x256) f32 -> f16 in WMMA
// B-fragment order:  dst = ((ktile*16 + ntile)*32 + lane)*16 + h
// with B element (k,n): k = ktile*32 + (lane>>4)*16 + h, n = ntile*16 + (lane&15)
// ---------------------------------------------------------------------------
#define UP_FRAGS    (4 * 16)              // KT=4, NT=16
#define DENSE_FRAGS (8 * 16)              // KT=8, NT=16
#define TOTAL_HALF  ((UP_FRAGS + 3 * DENSE_FRAGS) * 512)

__global__ void pack_weights_kernel(const float* __restrict__ W_up,
                                    const float* __restrict__ W_dense,
                                    _Float16* __restrict__ wpack) {
    int tid = blockIdx.x * 256 + threadIdx.x;
    if (tid >= TOTAL_HALF) return;
    int frag   = tid >> 9;                // fragment id (512 halves each)
    int within = tid & 511;
    int lane   = within >> 4;
    int h      = within & 15;
    int klocal = ((lane >> 4) << 4) + h;  // (lane>=16 ? 16 : 0) + h
    int nlocal = lane & 15;
    float v;
    if (frag < UP_FRAGS) {
        int ktile = frag >> 4, ntile = frag & 15;
        v = W_up[(ktile * 32 + klocal) * OUT_EMBED + ntile * 16 + nlocal];
    } else {
        int f = frag - UP_FRAGS;
        int layer = f >> 7, rem = f & 127;
        int ktile = rem >> 4, ntile = rem & 15;
        v = W_dense[layer * (OUT_EMBED * OUT_EMBED)
                    + (ktile * 32 + klocal) * OUT_EMBED + ntile * 16 + nlocal];
    }
    wpack[tid] = (_Float16)v;
}

// ---------------------------------------------------------------------------
// Kernel 3: per-edge gate + f64 scatter. One wave per edge; lane owns 4
// channels (float4 load of messages -> 512B per wave, coalesced).
// ---------------------------------------------------------------------------
__global__ __launch_bounds__(256)
void edge_gate_scatter_kernel(const float* __restrict__ messages,
                              const float* __restrict__ rbf,
                              const int*   __restrict__ idx_i,
                              const float* __restrict__ W_rbf,
                              double*      __restrict__ acc) {
    int wave = threadIdx.x >> 5;
    int lane = threadIdx.x & 31;
    int e = blockIdx.x * 8 + wave;
    if (e >= N_EDGES) return;

    float r[RBF_N];
#pragma unroll
    for (int j = 0; j < RBF_N; ++j) r[j] = rbf[(size_t)e * RBF_N + j]; // broadcast

    int ch = lane * 4;
    float4 g = make_float4(0.f, 0.f, 0.f, 0.f);
#pragma unroll
    for (int j = 0; j < RBF_N; ++j) {
        const float4 w = *(const float4*)(W_rbf + j * EMBED + ch);   // L1-resident
        g.x += r[j] * w.x; g.y += r[j] * w.y;
        g.z += r[j] * w.z; g.w += r[j] * w.w;
    }
    const float4 m = *(const float4*)(messages + (size_t)e * EMBED + ch);
    double* dst = acc + (size_t)idx_i[e] * EMBED + ch;
    atomic_add_f64(dst + 0, (double)(m.x * g.x));
    atomic_add_f64(dst + 1, (double)(m.y * g.y));
    atomic_add_f64(dst + 2, (double)(m.z * g.z));
    atomic_add_f64(dst + 3, (double)(m.w * g.w));
}

// ---------------------------------------------------------------------------
// Kernel 4: WMMA MLP. One wave per 16-row strip; A fragments gathered from
// LDS (two contiguous 16B runs per lane -> ds_load_b128), B fragments are
// contiguous 32B loads from packed weights (global_load_b128 pairs).
// MUST be force-inlined so InferAddressSpaces resolves LDS vs global.
// ---------------------------------------------------------------------------
template <int KT, bool ACT>
__device__ __forceinline__ void dense_layer_wmma(const _Float16* __restrict__ Wfrag,
                                                 const _Float16* Ain, _Float16* Aout,
                                                 const float* __restrict__ bias,
                                                 int lane) {
    const int row = lane & 15;
    const int grp = lane >> 4;
    const int width = KT * 32;

    v16h a[KT];
#pragma unroll
    for (int kt = 0; kt < KT; ++kt) {
        v16h t;
#pragma unroll
        for (int h = 0; h < 16; ++h) {
            int k = kt * 32 + grp * 16 + h;
            t[h] = Ain[row * width + k];                       // contiguous 16B runs
        }
        a[kt] = t;
    }

#pragma unroll
    for (int nt = 0; nt < 16; ++nt) {
        v8f c = {};
#pragma unroll
        for (int kt = 0; kt < KT; ++kt) {
            v16h b = *(const v16h*)(Wfrag + ((kt * 16 + nt) * 32 + lane) * 16);
            c = __builtin_amdgcn_wmma_f32_16x16x32_f16(
                    false, a[kt], false, b, (short)0, c, false, false);
        }
        int n = nt * 16 + (lane & 15);
        float bval = ACT ? bias[n] : 0.0f;
#pragma unroll
        for (int rr = 0; rr < 8; ++rr) {
            int mrow = grp * 8 + rr;
            float x = c[rr];
            if (ACT) { x += bval; x = x / (1.0f + __expf(-x)); }  // swish
            Aout[mrow * OUT_EMBED + n] = (_Float16)x;             // ds_store_b16
        }
    }
}

__global__ __launch_bounds__(64)
void mlp_wmma_kernel(const double*   __restrict__ acc,
                     const _Float16* __restrict__ wpack,
                     const float*    __restrict__ b_dense,
                     const float*    __restrict__ W_final,
                     float*          __restrict__ out) {
    __shared__ _Float16 smem[2][2][16 * OUT_EMBED];   // [wave][buf] = 32 KB
    const int wave = threadIdx.x >> 5;
    const int lane = threadIdx.x & 31;
    int strip = blockIdx.x * 2 + wave;
    if (strip >= STRIPS) strip = STRIPS - 1;          // duplicate tail, no exit
    const int row0 = strip * 16;

    _Float16* bufA = smem[wave][0];
    _Float16* bufB = smem[wave][1];

    // Stage f64 segment sums -> f16 activations (width 128)
    for (int i = lane; i < 16 * EMBED; i += 32)
        bufA[i] = (_Float16)(float)acc[(size_t)row0 * EMBED + i];

    const _Float16* W_up_p = wpack;
    const _Float16* Wd0 = wpack + UP_FRAGS * 512;
    const _Float16* Wd1 = Wd0 + DENSE_FRAGS * 512;
    const _Float16* Wd2 = Wd1 + DENSE_FRAGS * 512;

    dense_layer_wmma<4, false>(W_up_p, bufA, bufB, nullptr,                 lane);
    dense_layer_wmma<8, true >(Wd0,    bufB, bufA, b_dense + 0 * OUT_EMBED, lane);
    dense_layer_wmma<8, true >(Wd1,    bufA, bufB, b_dense + 1 * OUT_EMBED, lane);
    dense_layer_wmma<8, true >(Wd2,    bufB, bufA, b_dense + 2 * OUT_EMBED, lane);

    // Final projection [256 x 1]: lane pair (row, row+16) splits K in halves
    const int row  = lane & 15;
    const int koff = (lane >> 4) * 128;
    float s = 0.0f;
#pragma unroll 8
    for (int k = 0; k < 128; ++k)
        s += (float)bufA[row * OUT_EMBED + koff + k] * W_final[koff + k];
    s += __shfl_xor(s, 16);
    if (lane < 16) out[row0 + row] = s;
}

// ---------------------------------------------------------------------------
extern "C" void kernel_launch(void* const* d_in, const int* in_sizes, int n_in,
                              void* d_out, int out_size, void* d_ws, size_t ws_size,
                              hipStream_t stream) {
    const float* messages = (const float*)d_in[0];
    const float* rbf      = (const float*)d_in[1];
    const int*   idx_i    = (const int*)  d_in[2];
    //           d_in[3] = n_particles (scalar, fixed 50000)
    const float* W_rbf    = (const float*)d_in[4];
    const float* W_up     = (const float*)d_in[5];
    const float* W_dense  = (const float*)d_in[6];
    const float* b_dense  = (const float*)d_in[7];
    const float* W_final  = (const float*)d_in[8];
    float* out = (float*)d_out;

    double*   acc   = (double*)d_ws;                               // 51.2 MB
    _Float16* wpack = (_Float16*)((char*)d_ws + (size_t)N_ATOMS * EMBED * 8);

    // 1) zero f64 accumulator (6.4M doubles = 25000 * 256 exact)
    zero_acc_kernel<<<(N_ATOMS * EMBED) / 256, 256, 0, stream>>>(acc);

    // 2) pack weights into f16 fragment order
    pack_weights_kernel<<<(TOTAL_HALF + 255) / 256, 256, 0, stream>>>(W_up, W_dense, wpack);

    // 3) gate + f64 scatter (one wave per edge)
    edge_gate_scatter_kernel<<<(N_EDGES + 7) / 8, 256, 0, stream>>>(
        messages, rbf, idx_i, W_rbf, acc);

    // 4) WMMA MLP (2 strips per block)
    mlp_wmma_kernel<<<(STRIPS + 1) / 2, 64, 0, stream>>>(acc, wpack, b_dense, W_final, out);

    (void)in_sizes; (void)n_in; (void)out_size; (void)ws_size;
}
